// MultiHeadAttention_13194139533591
// MI455X (gfx1250) — compile-verified, matrix operand order checked by reading, not run on hardware
//
#include <hip/hip_runtime.h>

typedef __bf16 bf16;
typedef __attribute__((ext_vector_type(16))) __bf16 v16bf;
typedef __attribute__((ext_vector_type(8)))  __bf16 v8bf;
typedef __attribute__((ext_vector_type(8)))  float  v8f;
typedef __attribute__((ext_vector_type(4)))  int    v4i;

#define DEVINL __device__ __forceinline__

// ---- CDNA5 async global->LDS path (guarded; falls back to reg+ds_store) ----
#if defined(__has_builtin)
#  if __has_builtin(__builtin_amdgcn_global_load_async_to_lds_b128)
#    define HAS_ASYNC_LDS 1
#  endif
#endif
#ifndef HAS_ASYNC_LDS
#  define HAS_ASYNC_LDS 0
#endif

// Builtin signature (probe-verified): (v4i addrspace(1)* src, v4i addrspace(3)* dst,
// imm offset, imm cpol).
typedef __attribute__((address_space(1))) v4i gv4i;
typedef __attribute__((address_space(3))) v4i lv4i;

DEVINL void wait_async_all() {
#if HAS_ASYNC_LDS
#  if __has_builtin(__builtin_amdgcn_s_wait_asynccnt)
    __builtin_amdgcn_s_wait_asynccnt(0);
#  else
    asm volatile("s_wait_asynccnt 0x0" ::: "memory");
#  endif
#endif
}

// Copy ROWS x (CPR*8) bf16 tile (row-major, sstride elems) into padded LDS [ROWS][72].
// 16B chunks; 256 threads; compile-time trip count (no exec-mask branches).
template <int ROWS, int CPR>
DEVINL void stage_tile(const bf16* __restrict__ src, size_t sstride,
                       bf16 (*dst)[72], int tid) {
    constexpr int ITERS = (ROWS * CPR) / 256;
#pragma unroll
    for (int it = 0; it < ITERS; ++it) {
        const int c   = tid + it * 256;
        const int row = c / CPR;
        const int cc  = c % CPR;
        const bf16* s = src + (size_t)row * sstride + cc * 8;
        bf16* d = &dst[row][cc * 8];
#if HAS_ASYNC_LDS
        __builtin_amdgcn_global_load_async_to_lds_b128((gv4i*)s, (lv4i*)d, 0, 0);
#else
        *(v8bf*)d = *(const v8bf*)s;
#endif
    }
}

union FragU { v16bf f; struct { v8bf lo, hi; } h; };

// 16x32 bf16 WMMA fragment from a row-major matrix (A-layout; B-layout identical
// with "row" playing the N role).  lane<16: row=lane, K={0..7,16..23};
// lane>=16: row=lane-16, K={8..15,24..31}  => two contiguous 16B loads.
DEVINL v16bf load_frag(const bf16* base, int stride, int row, int col, int lane) {
    const int m  = lane & 15;
    const int ks = (lane & 16) ? 8 : 0;
    const bf16* p = base + (size_t)(row + m) * (size_t)stride + (col + ks);
    FragU u;
    u.h.lo = *reinterpret_cast<const v8bf*>(p);
    u.h.hi = *reinterpret_cast<const v8bf*>(p + 16);
    return u.f;
}

DEVINL v8f wmma_bf16(v16bf a, v16bf b, v8f c) {
    return __builtin_amdgcn_wmma_f32_16x16x32_bf16(false, a, false, b, (short)0, c, false, false);
}

// ---------------------------------------------------------------- converts
// 8 elements per thread: b128 in (x2), b128 out.
__global__ void k_f32_to_bf16(const float* __restrict__ in, bf16* __restrict__ out, int n) {
    int i = (blockIdx.x * blockDim.x + threadIdx.x) * 8;
    if (i >= n) return;
    const float4 A = *reinterpret_cast<const float4*>(in + i);
    const float4 B = *reinterpret_cast<const float4*>(in + i + 4);
    v8bf o;
    o[0] = (bf16)A.x; o[1] = (bf16)A.y; o[2] = (bf16)A.z; o[3] = (bf16)A.w;
    o[4] = (bf16)B.x; o[5] = (bf16)B.y; o[6] = (bf16)B.z; o[7] = (bf16)B.w;
    *reinterpret_cast<v8bf*>(out + i) = o;
}

// Tiled transpose: w [K,N] fp32 -> wt [N,K] bf16.  32x32 tile via LDS,
// coalesced on both sides.  block = (32,8), grid = (K/32, N/32).
__global__ void k_transpose_bf16(const float* __restrict__ w, bf16* __restrict__ wt,
                                 int K, int N) {
    __shared__ bf16 tile[32][33];
    const int k0 = blockIdx.x * 32;
    const int n0 = blockIdx.y * 32;
    const int tx = threadIdx.x;   // 0..31
    const int ty = threadIdx.y;   // 0..7
#pragma unroll
    for (int i = ty; i < 32; i += 8)
        tile[i][tx] = (bf16)w[(size_t)(k0 + i) * N + n0 + tx];   // coalesced over n
    __syncthreads();
#pragma unroll
    for (int i = ty; i < 32; i += 8)
        wt[(size_t)(n0 + i) * K + k0 + tx] = tile[tx][i];        // coalesced over k
}

// ---------------------------------------------------------------- GEMM mainloop
// Block tile 128M x 128N, 8 waves = 4(M) x 2(N); wave tile 32M x 64N.
// B (N-major [N][1024]) staged in LDS (async, ping-pong Bt[0]/Bt[1]).
// A fragments register-prefetched one k-step ahead (static double set a0a1/a2a3).
DEVINL void gemm_step(const bf16 (*Btc)[72], int wn, int lane,
                      v16bf a0, v16bf a1, v8f acc[2][4]) {
#pragma unroll
    for (int f = 0; f < 4; ++f) {
        v16bf b = load_frag(&Btc[wn * 64 + 16 * f][0], 72, 0, 0, lane);
        acc[0][f] = wmma_bf16(a0, b, acc[0][f]);
        acc[1][f] = wmma_bf16(a1, b, acc[1][f]);
    }
}

DEVINL void gemm_mainloop(const bf16* __restrict__ A, const bf16* __restrict__ Bn,
                          int mrow0, int nbase, int wn, int lane, int tid,
                          bf16 (*Bt)[128][72], v8f acc[2][4]) {
    const bf16* Bbase = Bn + (size_t)nbase * 1024;
    stage_tile<128, 4>(Bbase, 1024, Bt[0], tid);
    v16bf a0 = load_frag(A, 1024, mrow0,      0, lane);
    v16bf a1 = load_frag(A, 1024, mrow0 + 16, 0, lane);
    wait_async_all();
    __syncthreads();

    for (int k0 = 0; k0 < 1024; k0 += 64) {
        // ---- phase 0: compute k0 from Bt[0]/a0a1; prefetch k0+32 into Bt[1]/a2a3
        stage_tile<128, 4>(Bbase + (k0 + 32), 1024, Bt[1], tid);
        v16bf a2 = load_frag(A, 1024, mrow0,      k0 + 32, lane);
        v16bf a3 = load_frag(A, 1024, mrow0 + 16, k0 + 32, lane);
        gemm_step(Bt[0], wn, lane, a0, a1, acc);
        wait_async_all();
        __syncthreads();

        // ---- phase 1: compute k0+32 from Bt[1]/a2a3; prefetch k0+64 into Bt[0]/a0a1
        if (k0 + 64 < 1024) {
            stage_tile<128, 4>(Bbase + (k0 + 64), 1024, Bt[0], tid);
            a0 = load_frag(A, 1024, mrow0,      k0 + 64, lane);
            a1 = load_frag(A, 1024, mrow0 + 16, k0 + 64, lane);
        }
        gemm_step(Bt[1], wn, lane, a2, a3, acc);
        wait_async_all();
        __syncthreads();
    }
}

// ---------------------------------------------------------------- QKV GEMM
__global__ __launch_bounds__(256) void k_gemm_qkv(const bf16* __restrict__ xb,
                                                  const bf16* __restrict__ wt,
                                                  bf16* __restrict__ Qb,
                                                  bf16* __restrict__ Kb,
                                                  bf16* __restrict__ Vt) {
    __shared__ __align__(16) bf16 Bt[2][128][72];
    const int tid  = threadIdx.x;
    const int lane = tid & 31;
    const int wave = tid >> 5;
    const int wm = wave & 3, wn = wave >> 2;
    const int mbase = blockIdx.y * 128 + wm * 32;
    const int nbase = blockIdx.x * 128;

    v8f acc[2][4] = {};
    gemm_mainloop(xb, wt, mbase, nbase, wn, lane, tid, Bt, acc);

    const int nlo = lane & 15;
    const int mhi = (lane & 16) ? 8 : 0;
#pragma unroll
    for (int g = 0; g < 2; ++g)
#pragma unroll
        for (int f = 0; f < 4; ++f) {
            const int n   = nbase + wn * 64 + 16 * f + nlo;
            const int sec = n >> 10;          // 0=Q 1=K 2=V
            const int c   = n & 1023;
            const int h   = c >> 6;
            const int d   = c & 63;
#pragma unroll
            for (int r = 0; r < 8; ++r) {
                const int mrow = mbase + g * 16 + r + mhi;
                const int b    = mrow >> 11;
                const int t    = mrow & 2047;
                const int head = b * 16 + h;
                const float v  = acc[g][f][r];
                if (sec == 0)
                    Qb[((size_t)head * 2048 + t) * 64 + d] = (bf16)(v * 0.125f);
                else if (sec == 1)
                    Kb[((size_t)head * 2048 + t) * 64 + d] = (bf16)v;
                else
                    Vt[((size_t)head * 64 + d) * 2048 + t] = (bf16)v;
            }
        }
}

// ---------------------------------------------------------------- flash attention
// One block = one (head, 128-query slab); 8 waves x 16 query rows.
// K (64x64) and V^T (64x64) blocks staged in LDS once per workgroup, double-buffered.
__global__ __launch_bounds__(256) void k_attention(const bf16* __restrict__ Qb,
                                                   const bf16* __restrict__ Kb,
                                                   const bf16* __restrict__ Vt,
                                                   bf16* __restrict__ attn) {
    __shared__ __align__(16) bf16 Kt[2][64][72];
    __shared__ __align__(16) bf16 Vts[2][64][72];
    __shared__ __align__(16) bf16 ldsP[8][16][72];

    const int tid  = threadIdx.x;
    const int lane = tid & 31;
    const int wave = tid >> 5;
    const int bh   = blockIdx.x >> 4;    // b*16+h
    const int qblk = blockIdx.x & 15;
    const int q0   = qblk * 128 + wave * 16;

    const bf16* Qh = Qb + (size_t)bh * 2048 * 64;
    const bf16* Kh = Kb + (size_t)bh * 2048 * 64;
    const bf16* Vh = Vt + (size_t)bh * 64 * 2048;

    const v16bf qa0 = load_frag(Qh, 64, q0, 0,  lane);
    const v16bf qa1 = load_frag(Qh, 64, q0, 32, lane);

    v8f o[4] = {};
    float mrun[8], lrun[8];
    const float NEG_INF = -__builtin_inff();
#pragma unroll
    for (int r = 0; r < 8; ++r) { mrun[r] = NEG_INF; lrun[r] = 0.f; }

    const int nlo = lane & 15;
    const int mhi = (lane & 16) ? 8 : 0;
    const int ks  = (lane & 16) ? 8 : 0;

    const int kend = qblk * 128 + 128;   // block-uniform key range
    stage_tile<64, 8>(Kh, 64,   Kt[0],  tid);
    stage_tile<64, 8>(Vh, 2048, Vts[0], tid);
    wait_async_all();
    __syncthreads();

    int cur = 0;
    for (int kb = 0; kb < kend; kb += 64) {
        if (kb + 64 < kend) {
            stage_tile<64, 8>(Kh + (size_t)(kb + 64) * 64, 64,   Kt[cur ^ 1],  tid);
            stage_tile<64, 8>(Vh + (kb + 64),              2048, Vts[cur ^ 1], tid);
        }

        if (kb < q0 + 16) {   // wave-uniform causal cutoff
            // ---- S = Q @ K^T  (K fragments from LDS)
            v8f s[4] = {};
#pragma unroll
            for (int f = 0; f < 4; ++f) {
                v16bf b0 = load_frag(&Kt[cur][16 * f][0], 72, 0, 0,  lane);
                v16bf b1 = load_frag(&Kt[cur][16 * f][0], 72, 0, 32, lane);
                s[f] = wmma_bf16(qa0, b0, s[f]);
                s[f] = wmma_bf16(qa1, b1, s[f]);
            }

            if (kb + 63 > q0) {
#pragma unroll
                for (int f = 0; f < 4; ++f) {
                    const int kj = kb + 16 * f + nlo;
#pragma unroll
                    for (int r = 0; r < 8; ++r) {
                        const int qi = q0 + r + mhi;
                        if (kj > qi) s[f][r] = NEG_INF;
                    }
                }
            }

            // ---- online softmax (row = one VGPR across a 16-lane half)
#pragma unroll
            for (int r = 0; r < 8; ++r) {
                float mx = fmaxf(fmaxf(s[0][r], s[1][r]), fmaxf(s[2][r], s[3][r]));
                mx = fmaxf(mx, __shfl_xor(mx, 1, 32));
                mx = fmaxf(mx, __shfl_xor(mx, 2, 32));
                mx = fmaxf(mx, __shfl_xor(mx, 4, 32));
                mx = fmaxf(mx, __shfl_xor(mx, 8, 32));
                const float mnew = fmaxf(mrun[r], mx);
                const float corr = __expf(mrun[r] - mnew);
                float ps = 0.f;
#pragma unroll
                for (int f = 0; f < 4; ++f) {
                    const float p = __expf(s[f][r] - mnew);
                    s[f][r] = p;
                    ps += p;
                }
                ps += __shfl_xor(ps, 1, 32);
                ps += __shfl_xor(ps, 2, 32);
                ps += __shfl_xor(ps, 4, 32);
                ps += __shfl_xor(ps, 8, 32);
                lrun[r] = lrun[r] * corr + ps;
                mrun[r] = mnew;
#pragma unroll
                for (int f = 0; f < 4; ++f) o[f][r] *= corr;
            }

            // ---- stage P (bf16) in LDS: C-frag -> A-frag relayout
#pragma unroll
            for (int f = 0; f < 4; ++f)
#pragma unroll
                for (int r = 0; r < 8; ++r)
                    ldsP[wave][r + mhi][16 * f + nlo] = (bf16)s[f][r];

            __builtin_amdgcn_wave_barrier();
            asm volatile("s_wait_dscnt 0x0" ::: "memory");
            __builtin_amdgcn_wave_barrier();

            const bf16* pl = &ldsP[wave][nlo][0];
            FragU pu0, pu1;
            pu0.h.lo = *reinterpret_cast<const v8bf*>(pl + ks);
            pu0.h.hi = *reinterpret_cast<const v8bf*>(pl + ks + 16);
            pu1.h.lo = *reinterpret_cast<const v8bf*>(pl + 32 + ks);
            pu1.h.hi = *reinterpret_cast<const v8bf*>(pl + 32 + ks + 16);

            // ---- O += P @ V (V^T fragments from LDS)
#pragma unroll
            for (int f = 0; f < 4; ++f) {
                v16bf vb0 = load_frag(&Vts[cur][16 * f][0], 72, 0, 0,  lane);
                v16bf vb1 = load_frag(&Vts[cur][16 * f][0], 72, 0, 32, lane);
                o[f] = wmma_bf16(pu0.f, vb0, o[f]);
                o[f] = wmma_bf16(pu1.f, vb1, o[f]);
            }
        }

        wait_async_all();
        __syncthreads();
        cur ^= 1;
    }

    // ---- normalize, emit attn[b*2048+t][h*64+d] bf16
    const int b = bh >> 4, h = bh & 15;
#pragma unroll
    for (int f = 0; f < 4; ++f)
#pragma unroll
        for (int r = 0; r < 8; ++r) {
            const int t = q0 + r + mhi;
            const float v = o[f][r] / lrun[r];
            attn[((size_t)(b * 2048 + t)) * 1024 + h * 64 + 16 * f + nlo] = (bf16)v;
        }
}

// ---------------------------------------------------------------- out projection
__global__ __launch_bounds__(256) void k_gemm_out(const bf16* __restrict__ attn,
                                                  const bf16* __restrict__ wt,
                                                  float* __restrict__ out) {
    __shared__ __align__(16) bf16 Bt[2][128][72];
    const int tid  = threadIdx.x;
    const int lane = tid & 31;
    const int wave = tid >> 5;
    const int wm = wave & 3, wn = wave >> 2;
    const int mbase = blockIdx.y * 128 + wm * 32;
    const int nbase = blockIdx.x * 128;

    v8f acc[2][4] = {};
    gemm_mainloop(attn, wt, mbase, nbase, wn, lane, tid, Bt, acc);

    const int nlo = lane & 15;
    const int mhi = (lane & 16) ? 8 : 0;
#pragma unroll
    for (int g = 0; g < 2; ++g)
#pragma unroll
        for (int f = 0; f < 4; ++f) {
            const int n = nbase + wn * 64 + 16 * f + nlo;
#pragma unroll
            for (int r = 0; r < 8; ++r) {
                const int mrow = mbase + g * 16 + r + mhi;
                out[(size_t)mrow * 1024 + n] = acc[g][f][r];
            }
        }
}

// ---------------------------------------------------------------- launch
extern "C" void kernel_launch(void* const* d_in, const int* in_sizes, int n_in,
                              void* d_out, int out_size, void* d_ws, size_t ws_size,
                              hipStream_t stream) {
    const float* x     = (const float*)d_in[0];  // [4,2048,1024]
    const float* w_qkv = (const float*)d_in[1];  // [1024,3072]
    const float* w_out = (const float*)d_in[2];  // [1024,1024]
    float* out = (float*)d_out;
    char*  ws  = (char*)d_ws;

    constexpr size_t SZ_XB = (size_t)8192 * 1024 * 2;      // 16 MB (reused for attn output)
    constexpr size_t SZ_WQ = (size_t)3072 * 1024 * 2;      // 6 MB
    constexpr size_t SZ_WO = (size_t)1024 * 1024 * 2;      // 2 MB
    constexpr size_t SZ_Q  = (size_t)64 * 2048 * 64 * 2;   // 16 MB

    bf16* xb   = (bf16*)(ws);
    bf16* wqT  = (bf16*)(ws + SZ_XB);
    bf16* woT  = (bf16*)(ws + SZ_XB + SZ_WQ);
    bf16* Qb   = (bf16*)(ws + SZ_XB + SZ_WQ + SZ_WO);
    bf16* Kb   = (bf16*)(ws + SZ_XB + SZ_WQ + SZ_WO + SZ_Q);
    bf16* Vt   = (bf16*)(ws + SZ_XB + SZ_WQ + SZ_WO + 2 * SZ_Q);
    bf16* attn = xb;  // x is dead after the QKV GEMM; reuse its buffer

    k_f32_to_bf16<<<8388608 / 8 / 256, 256, 0, stream>>>(x, xb, 8388608);
    k_transpose_bf16<<<dim3(32, 96), dim3(32, 8), 0, stream>>>(w_qkv, wqT, 1024, 3072);
    k_transpose_bf16<<<dim3(32, 32), dim3(32, 8), 0, stream>>>(w_out, woT, 1024, 1024);
    k_gemm_qkv<<<dim3(24, 64), 256, 0, stream>>>(xb, wqT, Qb, Kb, Vt);
    k_attention<<<dim3(1024), 256, 0, stream>>>(Qb, Kb, Vt, attn);
    k_gemm_out<<<dim3(8, 64), 256, 0, stream>>>(attn, woT, out);
}
